// CondAttentionBlock_46093589021360
// MI455X (gfx1250) — matrix operand order repaired and checked by reading, not compiled, not run on hardware
//
#include <hip/hip_runtime.h>
#include <hip/hip_bf16.h>

typedef __attribute__((ext_vector_type(16))) __bf16 v16bf;
typedef __attribute__((ext_vector_type(8)))  float  v8f;
typedef int v4i_ld __attribute__((vector_size(16)));

#define BB 32
#define CC 256
#define LL 1024
#define HH 4
#define CH 64

// ---- async global->LDS probe (gfx1250 GLOBAL_LOAD_ASYNC_TO_LDS_B128, ASYNCcnt) ----
#if defined(__has_builtin)
# if __has_builtin(__builtin_amdgcn_global_load_async_to_lds_b128) && \
     __has_builtin(__builtin_amdgcn_s_wait_asynccnt)
#  define ASYNC_LDS 1
# endif
#endif
#ifndef ASYNC_LDS
# define ASYNC_LDS 0
#endif

// ---------- WMMA helpers ----------
__device__ __forceinline__ v8f wmma_bf16(v16bf a, v16bf b, v8f c) {
  return __builtin_amdgcn_wmma_f32_16x16x32_bf16(false, a, false, b, (short)0, c, false, false);
}
// A-matrix (16x32 bf16) K index for VGPR pair v (0..7), lane-half hv (0/1): 16*half + 2v
__device__ __forceinline__ int a_kbase(int v, int hv) {
  return 16 * hv + 2 * v;
}
// B-matrix (32x16 bf16) K index: 16*half + 2v

__device__ __forceinline__ float red16_max(float v) {
  v = fmaxf(v, __shfl_xor(v, 1, 32));
  v = fmaxf(v, __shfl_xor(v, 2, 32));
  v = fmaxf(v, __shfl_xor(v, 4, 32));
  v = fmaxf(v, __shfl_xor(v, 8, 32));
  return v;
}
__device__ __forceinline__ float red16_sum(float v) {
  v += __shfl_xor(v, 1, 32);
  v += __shfl_xor(v, 2, 32);
  v += __shfl_xor(v, 4, 32);
  v += __shfl_xor(v, 8, 32);
  return v;
}
__device__ __forceinline__ float red32_sum(float v) {
  v += __shfl_xor(v, 1, 32);  v += __shfl_xor(v, 2, 32);
  v += __shfl_xor(v, 4, 32);  v += __shfl_xor(v, 8, 32);
  v += __shfl_xor(v, 16, 32);
  return v;
}

// ---------- prep kernels ----------
__global__ void k_c01(const float* __restrict__ Wkv, const float* __restrict__ gw,
                      const float* __restrict__ gb, const float* __restrict__ bkv,
                      float* __restrict__ c1, float* __restrict__ c0) {
  int o = blockIdx.x * 256 + threadIdx.x;
  if (o >= 2 * CC) return;
  float s1 = 0.f, s0 = 0.f;
  const float* row = Wkv + (size_t)o * CC;
  for (int i = 0; i < CC; ++i) { float w = row[i]; s1 += w * gw[i]; s0 += w * gb[i]; }
  c1[o] = s1; c0[o] = s0 + bkv[o];
}

__global__ void k_cvt(const float* __restrict__ Wq, const float* __restrict__ Wp,
                      __bf16* __restrict__ Wqb, __bf16* __restrict__ Wpb) {
  int i = blockIdx.x * 256 + threadIdx.x;
  if (i < CC * CC) Wqb[i] = (__bf16)Wq[i];
  else if (i < 2 * CC * CC) Wpb[i - CC * CC] = (__bf16)Wp[i - CC * CC];
}

__global__ void k_audio(const float* __restrict__ emb, const float* __restrict__ Wa,
                        const float* __restrict__ ba, float* __restrict__ a) {
  int t = blockIdx.x * 256 + threadIdx.x;
  if (t >= BB * LL) return;
  int b = t >> 10, l = t & (LL - 1);
  const float* e = emb + (size_t)BB * 512 + (size_t)b * 512;
  const float* w = Wa + (size_t)l * 512;
  float s = 0.f;
  for (int k = 0; k < 512; ++k) s += e[k] * w[k];
  a[t] = s + ba[l];
}

__global__ void k_astats(float* __restrict__ a) {
  int b = blockIdx.x, tid = threadIdx.x;
  float* ab = a + (size_t)b * LL;
  float s = 0.f, s2 = 0.f;
  for (int i = tid; i < LL; i += 256) { float v = ab[i]; s += v; s2 += v * v; }
  s = red32_sum(s); s2 = red32_sum(s2);
  __shared__ float sh0[8], sh1[8];
  int wave = tid >> 5, lane = tid & 31;
  if (lane == 0) { sh0[wave] = s; sh1[wave] = s2; }
  __syncthreads();
  if (tid == 0) {
    float S = 0.f, S2 = 0.f;
    for (int w = 0; w < 8; ++w) { S += sh0[w]; S2 += sh1[w]; }
    float mu = S / (float)LL;
    float var = S2 / (float)LL - mu * mu;
    sh0[0] = mu; sh1[0] = rsqrtf(var + 1e-5f);
  }
  __syncthreads();
  float mu = sh0[0], rs = sh1[0];
  for (int i = tid; i < LL; i += 256) ab[i] = (ab[i] - mu) * rs;
}

__global__ void k_xstats(const float* __restrict__ x, const float* __restrict__ gw,
                         const float* __restrict__ gb, float* __restrict__ sx,
                         float* __restrict__ tx) {
  int bg = blockIdx.x, b = bg >> 5, g = bg & 31, tid = threadIdx.x;
  const float* xb = x + (((size_t)b * CC + g * 8) << 10);
  float s = 0.f, s2 = 0.f;
  for (int i = tid; i < 8192; i += 256) { float v = xb[i]; s += v; s2 += v * v; }
  s = red32_sum(s); s2 = red32_sum(s2);
  __shared__ float sh0[8], sh1[8];
  int wave = tid >> 5, lane = tid & 31;
  if (lane == 0) { sh0[wave] = s; sh1[wave] = s2; }
  __syncthreads();
  if (tid == 0) {
    float S = 0.f, S2 = 0.f;
    for (int w = 0; w < 8; ++w) { S += sh0[w]; S2 += sh1[w]; }
    float mu = S / 8192.f;
    float var = S2 / 8192.f - mu * mu;
    sh0[0] = mu; sh1[0] = rsqrtf(var + 1e-5f);
  }
  __syncthreads();
  if (tid < 8) {
    float mu = sh0[0], rs = sh1[0];
    int ch = g * 8 + tid;
    float w = gw[ch];
    float sc = rs * w;
    sx[b * CC + ch] = sc;
    tx[b * CC + ch] = gb[ch] - mu * sc;
  }
}

__global__ void k_xnorm(const float* __restrict__ x, const float* __restrict__ sx,
                        const float* __restrict__ tx, __bf16* __restrict__ xn) {
  size_t i = (size_t)blockIdx.x * 256 + threadIdx.x;
  int ch = (int)(i >> 10);
  xn[i] = (__bf16)(x[i] * sx[ch] + tx[ch]);
}

__global__ void k_kvfill(const float* __restrict__ an, const float* __restrict__ c1,
                         const float* __restrict__ c0, __bf16* __restrict__ qkv) {
  size_t i = (size_t)blockIdx.x * 256 + threadIdx.x;
  int l = (int)(i & (LL - 1));
  int o = (int)((i >> 10) & 511);
  int b = (int)(i >> 19);
  float v = an[b * LL + l] * c1[o] + c0[o];
  qkv[(((size_t)b * 768 + 256 + o) << 10) + l] = (__bf16)v;
}

// ---------- WMMA GEMM: q = Wq_bf @ xn + bq -> qkv rows 0..255 (bf16) ----------
__global__ __launch_bounds__(256) void k_gemm_q(const __bf16* __restrict__ Wb,
                                                const __bf16* __restrict__ Xn,
                                                const float* __restrict__ bias,
                                                __bf16* __restrict__ qkv) {
  int b = blockIdx.z;
  int wave = threadIdx.x >> 5, lane = threadIdx.x & 31;
  int hv = lane >> 4, nn = lane & 15, m = lane & 15;
  int m0 = blockIdx.y * 128 + wave * 16;
  int n0 = blockIdx.x * 64;
  const __bf16* Bm = Xn + ((size_t)b << 18);
  v8f acc[4];
#pragma unroll
  for (int r = 0; r < 8; ++r) {
    float bv = bias[m0 + r + 8 * hv];
    acc[0][r] = bv; acc[1][r] = bv; acc[2][r] = bv; acc[3][r] = bv;
  }
  for (int kk = 0; kk < CC; kk += 32) {
    v16bf av;
    const __bf16* arow = Wb + (size_t)(m0 + m) * CC + kk;
#pragma unroll
    for (int v = 0; v < 8; ++v) {
      int k = a_kbase(v, hv);
      av[2 * v] = arow[k]; av[2 * v + 1] = arow[k + 1];
    }
#pragma unroll
    for (int jt = 0; jt < 4; ++jt) {
      v16bf bvv;
      int col = n0 + jt * 16 + nn;
#pragma unroll
      for (int v = 0; v < 8; ++v) {
        int k = kk + hv * 16 + 2 * v;
        bvv[2 * v]     = Bm[((size_t)k << 10) + col];
        bvv[2 * v + 1] = Bm[((size_t)(k + 1) << 10) + col];
      }
      acc[jt] = wmma_bf16(av, bvv, acc[jt]);
    }
  }
  __bf16* qb = qkv + ((size_t)b * 768 << 10);
#pragma unroll
  for (int jt = 0; jt < 4; ++jt)
#pragma unroll
    for (int r = 0; r < 8; ++r) {
      int o = m0 + r + 8 * hv;
      int l = n0 + jt * 16 + nn;
      qb[((size_t)o << 10) + l] = (__bf16)acc[jt][r];
    }
}

// ---------- flash attention with async K/V staging ----------
__global__ __launch_bounds__(256) void k_attn(const __bf16* __restrict__ qkv,
                                              __bf16* __restrict__ att) {
  __shared__ __bf16 Kl[64 * 32];      // K chunk: 64 ch x 32 keys
  __shared__ __bf16 Vl[64 * 32];      // V chunk: 64 ch x 32 keys
  __shared__ __bf16 Pl[8 * 16 * 32];  // per-wave P staging (C->A relayout)
  int b = blockIdx.z, h = blockIdx.y;
  int tid = threadIdx.x;
  int wave = tid >> 5, lane = tid & 31;
  int hv = lane >> 4, nn = lane & 15, m = lane & 15;
  int t0 = blockIdx.x * 128 + wave * 16;
  const __bf16* qp = qkv + (((size_t)b * 768 + h * 192) << 10);
  const __bf16* kp = qp + ((size_t)64 << 10);
  const __bf16* vp = qp + ((size_t)128 << 10);

  // staging assignment: 256 threads cover 64 rows x 4 segments of 8 bf16 (16B)
  int srow = tid >> 2, sseg = tid & 3;

  // Q A-operands for K=64 (invariant across the key loop)
  v16bf aq[2];
#pragma unroll
  for (int s = 0; s < 2; ++s)
#pragma unroll
    for (int v = 0; v < 8; ++v) {
      int k = 32 * s + a_kbase(v, hv);
      aq[s][2 * v]     = qp[((size_t)k << 10) + t0 + m];
      aq[s][2 * v + 1] = qp[((size_t)(k + 1) << 10) + t0 + m];
    }

  float m_run[8], l_run[8];
#pragma unroll
  for (int r = 0; r < 8; ++r) { m_run[r] = -1e30f; l_run[r] = 0.f; }
  v8f outa[4];
#pragma unroll
  for (int jt = 0; jt < 4; ++jt)
#pragma unroll
    for (int r = 0; r < 8; ++r) outa[jt][r] = 0.f;

  const float s2 = 0.125f; // ch^{-1/2}, ch=64
  __bf16* pw = Pl + wave * 512;

  for (int s0 = 0; s0 < LL; s0 += 32) {
    // ---- stage K/V chunk into LDS (async DMA path on CDNA5) ----
    {
      const __bf16* gk = kp + ((size_t)srow << 10) + s0 + sseg * 8;
      const __bf16* gv = vp + ((size_t)srow << 10) + s0 + sseg * 8;
      __bf16* lk = &Kl[srow * 32 + sseg * 8];
      __bf16* lv = &Vl[srow * 32 + sseg * 8];
#if ASYNC_LDS
      __builtin_amdgcn_global_load_async_to_lds_b128((v4i_ld*)gk, (v4i_ld*)lk, 0, 0);
      __builtin_amdgcn_global_load_async_to_lds_b128((v4i_ld*)gv, (v4i_ld*)lv, 0, 0);
      __builtin_amdgcn_s_wait_asynccnt(0);
#else
      *(float4*)lk = *(const float4*)gk;
      *(float4*)lv = *(const float4*)gv;
#endif
    }
    __syncthreads();

    // ---- scores: 2 tiles of 16 keys, K = 64 channels, from Kl ----
    v8f sc[2];
#pragma unroll
    for (int st = 0; st < 2; ++st) {
#pragma unroll
      for (int r = 0; r < 8; ++r) sc[st][r] = 0.f;
#pragma unroll
      for (int ks = 0; ks < 2; ++ks) {
        v16bf bk;
#pragma unroll
        for (int v = 0; v < 8; ++v) {
          int c = ks * 32 + hv * 16 + 2 * v;
          bk[2 * v]     = Kl[c * 32 + st * 16 + nn];
          bk[2 * v + 1] = Kl[(c + 1) * 32 + st * 16 + nn];
        }
        sc[st] = wmma_bf16(aq[ks], bk, sc[st]);
      }
    }
    // ---- online softmax per query row ----
#pragma unroll
    for (int r = 0; r < 8; ++r) {
      float v0 = sc[0][r] * s2, v1 = sc[1][r] * s2;
      float rm = red16_max(fmaxf(v0, v1));
      float nm = fmaxf(m_run[r], rm);
      float al = __expf(m_run[r] - nm);
      float p0 = __expf(v0 - nm), p1 = __expf(v1 - nm);
      float rs = red16_sum(p0 + p1);
      l_run[r] = l_run[r] * al + rs;
      m_run[r] = nm;
#pragma unroll
      for (int jt = 0; jt < 4; ++jt) outa[jt][r] *= al;
      int mm = r + 8 * hv;
      pw[mm * 32 + nn]      = (__bf16)p0;
      pw[mm * 32 + 16 + nn] = (__bf16)p1;
    }
    // ---- reload P as an A operand (wave-local LDS) ----
    v16bf ap;
#pragma unroll
    for (int v = 0; v < 8; ++v) {
      int k = a_kbase(v, hv);
      ap[2 * v]     = pw[m * 32 + k];
      ap[2 * v + 1] = pw[m * 32 + k + 1];
    }
    // ---- P @ V from Vl: N = 64 channels (4 tiles), K = 32 keys ----
#pragma unroll
    for (int jt = 0; jt < 4; ++jt) {
      v16bf bv;
      int c = jt * 16 + nn;
#pragma unroll
      for (int v = 0; v < 8; ++v) {
        int k = hv * 16 + 2 * v;
        bv[2 * v]     = Vl[c * 32 + k];
        bv[2 * v + 1] = Vl[c * 32 + k + 1];
      }
      outa[jt] = wmma_bf16(ap, bv, outa[jt]);
    }
    __syncthreads(); // protect Kl/Vl before next chunk's staging
  }
  // ---- normalize and store att[b, 64h + c, t] ----
#pragma unroll
  for (int jt = 0; jt < 4; ++jt)
#pragma unroll
    for (int r = 0; r < 8; ++r) {
      float inv = 1.f / l_run[r];
      int c = h * CH + jt * 16 + nn;
      int t = t0 + r + 8 * hv;
      att[(((size_t)b * CC + c) << 10) + t] = (__bf16)(outa[jt][r] * inv);
    }
}

// ---------- WMMA GEMM: out = x + Wp_bf @ att + bp (f32 out) ----------
__global__ __launch_bounds__(256) void k_gemm_p(const __bf16* __restrict__ Wb,
                                                const __bf16* __restrict__ Att,
                                                const float* __restrict__ bias,
                                                const float* __restrict__ x,
                                                float* __restrict__ out) {
  int b = blockIdx.z;
  int wave = threadIdx.x >> 5, lane = threadIdx.x & 31;
  int hv = lane >> 4, nn = lane & 15, m = lane & 15;
  int m0 = blockIdx.y * 128 + wave * 16;
  int n0 = blockIdx.x * 64;
  const __bf16* Bm = Att + ((size_t)b << 18);
  v8f acc[4];
#pragma unroll
  for (int r = 0; r < 8; ++r) {
    float bv = bias[m0 + r + 8 * hv];
    acc[0][r] = bv; acc[1][r] = bv; acc[2][r] = bv; acc[3][r] = bv;
  }
  for (int kk = 0; kk < CC; kk += 32) {
    v16bf av;
    const __bf16* arow = Wb + (size_t)(m0 + m) * CC + kk;
#pragma unroll
    for (int v = 0; v < 8; ++v) {
      int k = a_kbase(v, hv);
      av[2 * v] = arow[k]; av[2 * v + 1] = arow[k + 1];
    }
#pragma unroll
    for (int jt = 0; jt < 4; ++jt) {
      v16bf bvv;
      int col = n0 + jt * 16 + nn;
#pragma unroll
      for (int v = 0; v < 8; ++v) {
        int k = kk + hv * 16 + 2 * v;
        bvv[2 * v]     = Bm[((size_t)k << 10) + col];
        bvv[2 * v + 1] = Bm[((size_t)(k + 1) << 10) + col];
      }
      acc[jt] = wmma_bf16(av, bvv, acc[jt]);
    }
  }
#pragma unroll
  for (int jt = 0; jt < 4; ++jt)
#pragma unroll
    for (int r = 0; r < 8; ++r) {
      int o = m0 + r + 8 * hv;
      int l = n0 + jt * 16 + nn;
      size_t idx = (((size_t)b * CC + o) << 10) + l;
      out[idx] = x[idx] + acc[jt][r];
    }
}

extern "C" void kernel_launch(void* const* d_in, const int* in_sizes, int n_in,
                              void* d_out, int out_size, void* d_ws, size_t ws_size,
                              hipStream_t stream) {
  const float* x       = (const float*)d_in[0];
  const float* emb     = (const float*)d_in[1];
  const float* W_audio = (const float*)d_in[2];
  const float* b_audio = (const float*)d_in[3];
  const float* gnx_w   = (const float*)d_in[4];
  const float* gnx_b   = (const float*)d_in[5];
  const float* gne_w   = (const float*)d_in[6];
  const float* gne_b   = (const float*)d_in[7];
  const float* Wq      = (const float*)d_in[8];
  const float* bq      = (const float*)d_in[9];
  const float* Wkv     = (const float*)d_in[10];
  const float* bkv     = (const float*)d_in[11];
  const float* Wp      = (const float*)d_in[12];
  const float* bp      = (const float*)d_in[13];
  float* out = (float*)d_out;

  char* wsp = (char*)d_ws;
  auto alloc = [&](size_t bytes) -> char* {
    char* p = wsp;
    wsp += (bytes + 255) & ~(size_t)255;
    return p;
  };
  float*  a_buf  = (float*)alloc((size_t)BB * LL * 4);
  float*  c1     = (float*)alloc(512 * 4);
  float*  c0     = (float*)alloc(512 * 4);
  float*  sx     = (float*)alloc((size_t)BB * CC * 4);
  float*  tx     = (float*)alloc((size_t)BB * CC * 4);
  __bf16* Wqb    = (__bf16*)alloc((size_t)CC * CC * 2);
  __bf16* Wpb    = (__bf16*)alloc((size_t)CC * CC * 2);
  __bf16* xn     = (__bf16*)alloc((size_t)BB * CC * LL * 2);
  __bf16* qkv    = (__bf16*)alloc((size_t)BB * 768 * LL * 2);
  __bf16* attb   = (__bf16*)alloc((size_t)BB * CC * LL * 2);

  k_c01<<<2, 256, 0, stream>>>(Wkv, gne_w, gne_b, bkv, c1, c0);
  k_cvt<<<512, 256, 0, stream>>>(Wq, Wp, Wqb, Wpb);
  k_audio<<<(BB * LL) / 256, 256, 0, stream>>>(emb, W_audio, b_audio, a_buf);
  k_astats<<<BB, 256, 0, stream>>>(a_buf);
  k_xstats<<<BB * 32, 256, 0, stream>>>(x, gnx_w, gnx_b, sx, tx);
  k_xnorm<<<(BB * CC * LL) / 256, 256, 0, stream>>>(x, sx, tx, xn);
  k_kvfill<<<(BB * 512 * LL) / 256, 256, 0, stream>>>(a_buf, c1, c0, qkv);

  dim3 ggrid(LL / 64, CC / 128, BB); // 16 x 2 x 32
  k_gemm_q<<<ggrid, 256, 0, stream>>>(Wqb, xn, bq, qkv);

  dim3 agrid(LL / 128, HH, BB); // 8 x 4 x 32
  k_attn<<<agrid, 256, 0, stream>>>(qkv, attb);

  k_gemm_p<<<ggrid, 256, 0, stream>>>(Wpb, attb, bp, x, out);
}
